// MultiHeadAttentionLayer_wMem_13271448945373
// MI455X (gfx1250) — compile-verified, matrix operand order checked by reading, not run on hardware
//
#include <hip/hip_runtime.h>

// ---------------------------------------------------------------------------
// MultiHeadAttention with memory slots, bf16 WMMA pipeline for gfx1250.
// B=2, S=2048, D=1024, H=16, HD=64, M=64 memory slots, T = S+M = 2112 keys.
// ---------------------------------------------------------------------------

typedef __attribute__((ext_vector_type(16))) __bf16 v16bf;
typedef __attribute__((ext_vector_type(8)))  float  v8f;
typedef __attribute__((ext_vector_type(4)))  unsigned int v4u;
typedef __attribute__((ext_vector_type(8)))  int    v8i;
typedef __attribute__((ext_vector_type(4)))  int    v4i;

#define B_  2
#define S_  2048
#define D_  1024
#define H_  16
#define HD_ 64
#define M_  64
#define T_  2112   // S_ + M_

// D = A x B + C, bf16 16x16x32, f32 accum
__device__ __forceinline__ v8f wmma_bf16(v16bf a, v16bf b, v8f c) {
  return __builtin_amdgcn_wmma_f32_16x16x32_bf16(
      /*neg_a=*/false, a, /*neg_b=*/false, b,
      /*c_mod=*/(short)0, c, /*reuse_a=*/false, /*reuse_b=*/false);
}

// Per-lane fragment chunks (A: lane = M row, B: lane = N row of W / K / Vt):
//   halfwords 0..7  -> k = k0 + 8*g + {0..7}
//   halfwords 8..15 -> k = k0 + 8*g + 16 + {0..7}    (g = lane>>4)
// p points at row_base + k0.
__device__ __forceinline__ v16bf load_frag_f32(const float* p, int g) {
  const float4* c0 = (const float4*)(p + 8 * g);
  const float4* c1 = (const float4*)(p + 8 * g + 16);
  float4 a0 = c0[0], a1 = c0[1];
  float4 b0 = c1[0], b1 = c1[1];
  float t[16] = {a0.x, a0.y, a0.z, a0.w, a1.x, a1.y, a1.z, a1.w,
                 b0.x, b0.y, b0.z, b0.w, b1.x, b1.y, b1.z, b1.w};
  v16bf r;
#pragma unroll
  for (int i = 0; i < 16; ++i) r[i] = (__bf16)t[i];
  return r;
}

__device__ __forceinline__ v16bf load_frag_bf16(const __bf16* p, int g) {
  v16bf r;
  ((uint4*)&r)[0] = *(const uint4*)(p + 8 * g);        // 8 bf16 = 16B
  ((uint4*)&r)[1] = *(const uint4*)(p + 8 * g + 16);
  return r;
}

// ---- DPP16 butterfly reduction within each 16-lane half ---------------------
// Stages: quad_perm xor1 (0xB1), quad_perm xor2 (0x4E),
//         row_half_mirror xor7 (0x141), row_mirror xor15 (0x140).
// Each stage combines disjoint partial sets -> valid for max AND sum.
template <int CTRL>
__device__ __forceinline__ float dpp_xor(float x) {
  const int v = __float_as_int(x);
  const int r = __builtin_amdgcn_update_dpp(v, v, CTRL, 0xf, 0xf, false);
  return __int_as_float(r);
}
template <int CTRL>
__device__ __forceinline__ void stage_max(v8f& v) {
#pragma unroll
  for (int r = 0; r < 8; ++r) v[r] = fmaxf(v[r], dpp_xor<CTRL>(v[r]));
}
template <int CTRL>
__device__ __forceinline__ void stage_sum(v8f& v) {
#pragma unroll
  for (int r = 0; r < 8; ++r) v[r] += dpp_xor<CTRL>(v[r]);
}
__device__ __forceinline__ void reduce16_max(v8f& v) {
  stage_max<0xB1>(v); stage_max<0x4E>(v);
  stage_max<0x141>(v); stage_max<0x140>(v);
}
__device__ __forceinline__ void reduce16_sum(v8f& v) {
  stage_sum<0xB1>(v); stage_sum<0x4E>(v);
  stage_sum<0x141>(v); stage_sum<0x140>(v);
}

// ---------------------------------------------------------------------------
// C[row, col] = (A[row,:] . W[col,:]) + bias[col], scaled; optional bf16 out,
// optional transposed output (for V^T), with per-batch output row remapping
// so K/V land inside the [B, 2112, D] concat layout.
// WG = 256 threads = 8 waves as 4x2 -> 128x64 tile; wave tile 32x32.
// ---------------------------------------------------------------------------
template <bool A_IS_BF16, bool OUT_BF16, bool OUT_TRANS>
__global__ __launch_bounds__(256) void gemm_bt(
    const void* __restrict__ Av, const float* __restrict__ W,
    const float* __restrict__ bias, void* __restrict__ Cv,
    int K, int N, int rowsInPerBatch, int rowsOutPerBatch, int ldc,
    float out_scale) {
  const int wave = threadIdx.x >> 5;
  const int lane = threadIdx.x & 31;
  const int wm = wave & 3, wn = wave >> 2;
  const int m0 = blockIdx.x * 128 + wm * 32;
  const int n0 = blockIdx.y * 64 + wn * 32;
  const int g = lane >> 4;
  const int lr = lane & 15;

  const float*  A_f = (const float*)Av;
  const __bf16* A_b = (const __bf16*)Av;

  v8f acc[2][2] = {};

  for (int k0 = 0; k0 < K; k0 += 32) {
    v16bf a[2], b[2];
#pragma unroll
    for (int i = 0; i < 2; ++i) {
      const int row = m0 + i * 16 + lr;
      if (A_IS_BF16)
        a[i] = load_frag_bf16(A_b + (size_t)row * K + k0, g);
      else
        a[i] = load_frag_f32(A_f + (size_t)row * K + k0, g);
      const int col = n0 + i * 16 + lr;
      b[i] = load_frag_f32(W + (size_t)col * K + k0, g);
      __builtin_prefetch(W + (size_t)col * K + k0 + 64, 0, 1);  // global_prefetch
    }
#pragma unroll
    for (int i = 0; i < 2; ++i)
#pragma unroll
      for (int j = 0; j < 2; ++j) acc[i][j] = wmma_bf16(a[i], b[j], acc[i][j]);
  }

  // Epilogue: C layout -> col = lr (+16j), row = vgpr r + 8*g (+16i)
#pragma unroll
  for (int i = 0; i < 2; ++i)
#pragma unroll
    for (int j = 0; j < 2; ++j) {
      const int col = n0 + j * 16 + lr;
      const float bval = bias[col];
#pragma unroll
      for (int r = 0; r < 8; ++r) {
        const int row = m0 + i * 16 + g * 8 + r;
        const float v = (acc[i][j][r] + bval) * out_scale;
        const int bidx = row / rowsInPerBatch;
        const int rIn  = row % rowsInPerBatch;
        const size_t idx =
            OUT_TRANS
                ? ((size_t)(bidx * N + col)) * rowsOutPerBatch + rIn
                : ((size_t)bidx * rowsOutPerBatch + rIn) * (size_t)ldc + col;
        if (OUT_BF16)
          ((__bf16*)Cv)[idx] = (__bf16)v;
        else
          ((float*)Cv)[idx] = v;
      }
    }
}

// ---------------------------------------------------------------------------
// Fill memory-slot rows: Kbf[b][2048+j][d] = 8*m_k[j][d]  (sqrt(HD)=8)
//                        Vt [b][d][2048+j] = 8*m_v[j][d]  (sqrt(M)=8)
// ---------------------------------------------------------------------------
__global__ void fill_mem(const float* __restrict__ m_k,
                         const float* __restrict__ m_v,
                         __bf16* __restrict__ Kbf, __bf16* __restrict__ Vt) {
  const int idx = blockIdx.x * blockDim.x + threadIdx.x;  // over M_*D_
  if (idx >= M_ * D_) return;
  const int j = idx / D_, d = idx % D_;
  const __bf16 kv = (__bf16)(8.0f * m_k[idx]);
  const __bf16 vv = (__bf16)(8.0f * m_v[idx]);
#pragma unroll
  for (int b = 0; b < B_; ++b) {
    Kbf[((size_t)(b * T_ + S_ + j)) * D_ + d] = kv;
    Vt[((size_t)(b * D_ + d)) * T_ + S_ + j] = vv;
  }
}

// ---------------------------------------------------------------------------
// Flash attention over 2112 keys (last 64 = memory slots, never masked).
// Q is pre-scaled by 1/sqrt(HD). 128 threads = 4 waves, 16 q-rows per wave.
// Mask is staged into LDS via the Tensor Data Mover (1-row 2048x4B tile),
// with a plain staging loop afterwards as a correctness backstop.
// ---------------------------------------------------------------------------
__global__ __launch_bounds__(128) void attn(
    const __bf16* __restrict__ Qbf, const __bf16* __restrict__ Kbf,
    const __bf16* __restrict__ Vt, const int* __restrict__ mask,
    __bf16* __restrict__ Xbf) {
  __shared__ int smask[S_];
  __shared__ __bf16 pbuf[4][16][40];  // 80B row stride -> 16B-aligned chunks

  const int bh = blockIdx.y;
  const int b = bh >> 4, h = bh & 15;
  const int wave = threadIdx.x >> 5;
  const int lane = threadIdx.x & 31;
  const int g = lane >> 4;
  const int lr = lane & 15;
  const int q0 = blockIdx.x * 64 + wave * 16;

  // ---- TDM: async DMA of mask[b, 0:2048] into smask (one wave issues) -----
  if (wave == 0) {
    const unsigned long long ga =
        (unsigned long long)(uintptr_t)(mask + (size_t)b * S_);
    const unsigned int lds = (unsigned int)(uintptr_t)&smask[0];  // LDS offset
    // D# group0: count=1 | lds_addr | global_addr[56:0] | type=2
    v4u g0 = {1u, lds, (unsigned int)ga,
              ((unsigned int)(ga >> 32) & 0x01FFFFFFu) | 0x80000000u};
    // D# group1: data_size=4B; tensor_dim0=2048; tensor_dim1=1;
    //            tile_dim0=2048; tile_dim1=1; tensor_dim0_stride=2048
    v8i g1 = {(int)(2u << 16),     // wg_mask=0, data_size=2 (4B)
              (int)(2048u << 16),  // tensor_dim0[15:0] @ bits[63:48]
              (int)(1u << 16),     // tensor_dim1[15:0] @ bits[95:80]
              (int)(2048u << 16),  // tile_dim0 @ bits[127:112]
              1,                   // tile_dim1=1, tile_dim2=0
              2048,                // tensor_dim0_stride[31:0]
              0, 0};
    v4i g2 = {0, 0, 0, 0};
    v4i g3 = {0, 0, 0, 0};
    v8i g4 = {0, 0, 0, 0, 0, 0, 0, 0};  // 6-arg toolchain variant: extra group
    __builtin_amdgcn_tensor_load_to_lds(g0, g1, g2, g3, g4, 0);
    __builtin_amdgcn_s_wait_tensorcnt(0);
  }
  __syncthreads();
  // Backstop: rewrite smask with known-good values (cheap, once per WG).
  for (int i = threadIdx.x; i < S_; i += 128) smask[i] = mask[b * S_ + i];
  __syncthreads();

  // Q fragments for this wave's 16 rows (held for the whole kernel)
  const __bf16* qbase = Qbf + ((size_t)(b * S_ + q0 + lr)) * D_ + h * HD_;
  const v16bf qf0 = load_frag_bf16(qbase, g);       // hd 0..31
  const v16bf qf1 = load_frag_bf16(qbase + 32, g);  // hd 32..63

  const __bf16* kbhead = Kbf + ((size_t)b * T_) * D_ + h * HD_;
  const __bf16* vbhead = Vt + ((size_t)(b * D_ + h * HD_)) * T_;

  v8f o[4] = {};
  float mrun[8], lrun[8];
#pragma unroll
  for (int r = 0; r < 8; ++r) { mrun[r] = -3.0e38f; lrun[r] = 0.0f; }

  for (int t0 = 0; t0 < T_; t0 += 32) {
    // ---- scores: S[16q x 32k] as two 16x16 accumulators -------------------
    v8f s0 = {}, s1 = {};
    {
      const __bf16* kr0 = kbhead + (size_t)(t0 + lr) * D_;
      const __bf16* kr1 = kbhead + (size_t)(t0 + 16 + lr) * D_;
      s0 = wmma_bf16(qf0, load_frag_bf16(kr0, g), s0);
      s0 = wmma_bf16(qf1, load_frag_bf16(kr0 + 32, g), s0);
      s1 = wmma_bf16(qf0, load_frag_bf16(kr1, g), s1);
      s1 = wmma_bf16(qf1, load_frag_bf16(kr1 + 32, g), s1);
    }

    // ---- mask (branchless; memory slots t>=S_ are never masked) -----------
    const int c0 = t0 + lr, c1 = c0 + 16;
    const bool kill0 = (c0 < S_) && (smask[c0] == 0);
    const bool kill1 = (c1 < S_) && (smask[c1] == 0);
#pragma unroll
    for (int r = 0; r < 8; ++r) {
      s0[r] = kill0 ? -1.0e20f : s0[r];
      s1[r] = kill1 ? -1.0e20f : s1[r];
    }

    // ---- row max over 32 keys (DPP butterfly within 16-lane half) ---------
    v8f sm;
#pragma unroll
    for (int r = 0; r < 8; ++r) sm[r] = fmaxf(s0[r], s1[r]);
    reduce16_max(sm);

    // ---- online softmax update -------------------------------------------
    float corr[8];
#pragma unroll
    for (int r = 0; r < 8; ++r) {
      const float mn = fmaxf(mrun[r], sm[r]);
      corr[r] = __expf(mrun[r] - mn);
      mrun[r] = mn;
    }
    v8f p0, p1, rsum;
#pragma unroll
    for (int r = 0; r < 8; ++r) {
      p0[r] = __expf(s0[r] - mrun[r]);
      p1[r] = __expf(s1[r] - mrun[r]);
      rsum[r] = p0[r] + p1[r];
    }
    reduce16_sum(rsum);
#pragma unroll
    for (int r = 0; r < 8; ++r) lrun[r] = lrun[r] * corr[r] + rsum[r];
#pragma unroll
    for (int j = 0; j < 4; ++j)
#pragma unroll
      for (int r = 0; r < 8; ++r) o[j][r] *= corr[r];

    // ---- transpose P through LDS into A-fragment layout -------------------
    __syncthreads();  // WAR vs previous iteration's fragment reads
#pragma unroll
    for (int r = 0; r < 8; ++r) {
      pbuf[wave][r + 8 * g][lr]      = (__bf16)p0[r];
      pbuf[wave][r + 8 * g][lr + 16] = (__bf16)p1[r];
    }
    __syncthreads();
    v16bf pf;
    ((uint4*)&pf)[0] = *(const uint4*)&pbuf[wave][lr][8 * g];        // ds_load_b128
    ((uint4*)&pf)[1] = *(const uint4*)&pbuf[wave][lr][8 * g + 16];

    // ---- O += P (16x32) @ V (32x64), V^T columns are contiguous -----------
#pragma unroll
    for (int j = 0; j < 4; ++j) {
      const __bf16* vcol = vbhead + (size_t)(j * 16 + lr) * T_ + t0;
      o[j] = wmma_bf16(pf, load_frag_bf16(vcol, g), o[j]);
    }
  }

  // ---- finalize: divide by row sums, write bf16 [B*S, D] -------------------
  float inv[8];
#pragma unroll
  for (int r = 0; r < 8; ++r) inv[r] = 1.0f / lrun[r];
#pragma unroll
  for (int j = 0; j < 4; ++j)
#pragma unroll
    for (int r = 0; r < 8; ++r) {
      const int qrow = q0 + r + 8 * g;
      const int col = h * HD_ + j * 16 + lr;
      Xbf[((size_t)(b * S_ + qrow)) * D_ + col] = (__bf16)(o[j][r] * inv[r]);
    }
}

// ---------------------------------------------------------------------------
extern "C" void kernel_launch(void* const* d_in, const int* in_sizes, int n_in,
                              void* d_out, int out_size, void* d_ws,
                              size_t ws_size, hipStream_t stream) {
  const float* query = (const float*)d_in[0];
  const float* key   = (const float*)d_in[1];
  const float* value = (const float*)d_in[2];
  const int*   mask  = (const int*)d_in[3];
  const float* Wq = (const float*)d_in[4];
  const float* bq = (const float*)d_in[5];
  const float* Wk = (const float*)d_in[6];
  const float* bk = (const float*)d_in[7];
  const float* Wv = (const float*)d_in[8];
  const float* bv = (const float*)d_in[9];
  const float* Wo = (const float*)d_in[10];
  const float* bo = (const float*)d_in[11];
  const float* m_k = (const float*)d_in[12];
  const float* m_v = (const float*)d_in[13];

  char* ws = (char*)d_ws;
  __bf16* Qbf = (__bf16*)ws; ws += (size_t)B_ * S_ * D_ * 2;  // [B*S, D], pre-scaled
  __bf16* Kbf = (__bf16*)ws; ws += (size_t)B_ * T_ * D_ * 2;  // [B*T, D]
  __bf16* Vt  = (__bf16*)ws; ws += (size_t)B_ * D_ * T_ * 2;  // [B, D, T]
  __bf16* Xbf = (__bf16*)ws;                                  // [B*S, D]

  const dim3 gg(B_ * S_ / 128, D_ / 64);  // (32, 16)
  const dim3 gb(256);

  // Projections (fp32 in -> bf16 out via WMMA); Q pre-scaled by 1/sqrt(HD)
  gemm_bt<false, true, false><<<gg, gb, 0, stream>>>(
      query, Wq, bq, Qbf, D_, D_, S_, S_, D_, 0.125f);
  gemm_bt<false, true, false><<<gg, gb, 0, stream>>>(
      key, Wk, bk, Kbf, D_, D_, S_, T_, D_, 1.0f);
  gemm_bt<false, true, true><<<gg, gb, 0, stream>>>(
      value, Wv, bv, Vt, D_, D_, S_, T_, D_, 1.0f);

  fill_mem<<<(M_ * D_ + 255) / 256, 256, 0, stream>>>(m_k, m_v, Kbf, Vt);

  const dim3 ga(S_ / 64, B_ * H_);  // (32, 32)
  attn<<<ga, dim3(128), 0, stream>>>(Qbf, Kbf, Vt, mask, Xbf);

  // Output projection: bf16 X @ Wo^T + bo -> fp32 d_out
  gemm_bt<true, false, false><<<gg, gb, 0, stream>>>(
      Xbf, Wo, bo, (float*)d_out, D_, D_, S_, S_, D_, 1.0f);
}
